// Model_73083163509008
// MI455X (gfx1250) — matrix operand order repaired and checked by reading, not compiled, not run on hardware
//
#include <hip/hip_runtime.h>
#include <math.h>

// ---------------- model constants ----------------
#define LTOK   1520
#define NB     4
#define DMODEL 512
#define NHEAD  8
#define DHEAD  64
#define DFFN   2048
#define SK     40
#define TLAG   12
#define MROWS  (NB * LTOK)        // 6080
#define QKVLD  (3 * DMODEL)       // 1536

// ---------------- WMMA types ----------------
typedef __bf16 v16bf __attribute__((ext_vector_type(16)));
typedef __bf16 v8bf  __attribute__((ext_vector_type(8)));
typedef float  v8f   __attribute__((ext_vector_type(8)));
typedef float  v8ff  __attribute__((ext_vector_type(8)));

// convert 8 floats -> 8 bf16 (native v_cvt_pk_bf16_f32 path) and store 16B to LDS
__device__ inline void cvt_store8(__bf16* dst, v8ff f) {
    *(v8bf*)dst = __builtin_convertvector(f, v8bf);
}

__device__ inline float gelu_f(float x) {
    float x3 = x * x * x;
    return 0.5f * x * (1.0f + tanhf(0.7978845608f * (x + 0.044715f * x3)));
}

// ---------------- generic WMMA GEMM ----------------
// C[M,N] = act( scale * (A[M,K] @ B) + bias ), B row-major [K,N] (transB=0)
// or B = W[N,K] row-major interpreted as [K,N] (transB=1).
// Block tile 64x128; 8 waves = 4(M) x 2(N); wave = 16x64 via 4 WMMA sharing A frag.
// LDS tiles are pre-swizzled into WMMA fragment order (32B contiguous per lane).
// EXACT=1: requires M%64==0, N%128==0, K%32==0 (no guards, vector loads).
// EXACT=0: requires only M%16==0, N%16==0 (true for every call in this model);
//          loads use clamped addresses + value select (no exec branching).
#define BM 64
#define BN 128
#define BK 32

template<int EXACT>
__global__ __launch_bounds__(256)
void gemm_wmma(const float* __restrict__ A, int lda,
               const float* __restrict__ Wt, int ldw, int transB,
               const float* __restrict__ bias,
               float* __restrict__ C, int ldc,
               int M, int N, int K, int act, float scale)
{
    __shared__ __align__(32) __bf16 Afrag[4][32][16];  // 4 KB
    __shared__ __align__(32) __bf16 Bfrag[8][32][16];  // 8 KB

    const int tid  = threadIdx.x;
    const int lane = tid & 31;
    const int wave = tid >> 5;
    const int wm   = wave & 3;   // 0..3 : M sub-tile (16 rows)
    const int wn   = wave >> 2;  // 0..1 : N sub-tile (64 cols)
    const int m0   = blockIdx.y * BM;
    const int n0   = blockIdx.x * BN;

    v8f acc[4] = {{}, {}, {}, {}};

    // A staging constants (one K-octet per thread)
    const int ar  = tid >> 2;            // 0..63 tile row
    const int aoc = tid & 3;             // K octet
    const int ac0 = aoc << 3;
    const int alh = aoc & 1;
    const int ajb = (aoc >> 1) << 3;
    __bf16* adst = &Afrag[ar >> 4][(ar & 15) + (alh << 4)][ajb];

    for (int kk = 0; kk < K; kk += BK) {
        // ---- stage A tile (64x32) ----
        {
            v8ff f;
            if (EXACT) {
                f = *(const v8ff*)(A + (long)(m0 + ar) * lda + kk + ac0);
            } else {
                int gr  = m0 + ar;
                int grc = gr < M ? gr : M - 1;
                bool rok = gr < M;
                const float* base = A + (long)grc * lda;
#pragma unroll
                for (int i = 0; i < 8; ++i) {
                    int gc  = kk + ac0 + i;
                    int gcc = gc < K ? gc : K - 1;
                    float v = base[gcc];
                    f[i] = (rok && gc < K) ? v : 0.0f;
                }
            }
            cvt_store8(adst, f);
        }
        // ---- stage B tile (32x128): 512 units (col x K-octet), 2 per thread ----
#pragma unroll
        for (int uu = 0; uu < 2; ++uu) {
            int u  = tid + uu * 256;
            int c  = u >> 2;          // 0..127 tile col
            int ko = u & 3;           // K octet
            int k0 = ko << 3;
            int kh = ko & 1;
            int jb = (ko >> 1) << 3;
            int gn = n0 + c;
            v8ff f;
            if (EXACT) {
                if (transB) {
                    f = *(const v8ff*)(Wt + (long)gn * ldw + kk + k0);
                } else {
                    const float* src = Wt + (long)(kk + k0) * ldw + gn;
#pragma unroll
                    for (int i = 0; i < 8; ++i) f[i] = src[(long)i * ldw];
                }
            } else {
                int gnc = gn < N ? gn : N - 1;
                bool cok = gn < N;
#pragma unroll
                for (int i = 0; i < 8; ++i) {
                    int gk  = kk + k0 + i;
                    int gkc = gk < K ? gk : K - 1;
                    float v = transB ? Wt[(long)gnc * ldw + gkc]
                                     : Wt[(long)gkc * ldw + gnc];
                    f[i] = (cok && gk < K) ? v : 0.0f;
                }
            }
            cvt_store8(&Bfrag[c >> 4][(c & 15) + (kh << 4)][jb], f);
        }
        __syncthreads();

        // contiguous 32B fragment loads
        v16bf a = *(const v16bf*)&Afrag[wm][lane][0];
#pragma unroll
        for (int t = 0; t < 4; ++t) {
            v16bf b = *(const v16bf*)&Bfrag[wn * 4 + t][lane][0];
            acc[t] = __builtin_amdgcn_wmma_f32_16x16x32_bf16(false, a, false, b, (short)0, acc[t], false, false);
        }
        __syncthreads();
    }

    // Store. M,N are multiples of 16 in all uses -> tile validity is uniform.
    if (EXACT || (m0 + wm * 16) < M) {
        int rbase = m0 + wm * 16 + ((lane >> 4) << 3);
#pragma unroll
        for (int t = 0; t < 4; ++t) {
            int ct0 = n0 + wn * 64 + t * 16;
            if (!EXACT && ct0 >= N) continue;
            int col = ct0 + (lane & 15);
            float bv = bias ? bias[col] : 0.0f;
#pragma unroll
            for (int r = 0; r < 8; ++r) {
                float v = acc[t][r] * scale + bv;
                if (act == 1) v = gelu_f(v); else if (act == 2) v = fmaxf(v, 0.0f);
                C[(long)(rbase + r) * ldc + col] = v;
            }
        }
    }
}

// ---------------- positional encoding + input gather ----------------
__global__ void pe_kernel(float* __restrict__ pe) {
    int t = blockIdx.x * blockDim.x + threadIdx.x;
    if (t >= LTOK * DMODEL) return;
    int d = t % DMODEL, l = t / DMODEL;
    float expo = (float)(2 * (d / 2)) / (float)DMODEL;
    float ang  = (float)l / powf(10000.0f, expo);
    pe[t] = (d & 1) ? cosf(ang) : sinf(ang);
}

__global__ void build_x(const float* __restrict__ inflow, float* __restrict__ x) {
    int t = blockIdx.x * blockDim.x + threadIdx.x;
    if (t >= MROWS * TLAG) return;
    int c = t % TLAG;
    int l = (t / TLAG) % LTOK;
    int b = t / (TLAG * LTOK);
    int flat = l * TLAG + c;
    int tt = flat / LTOK, rest = flat % LTOK;
    x[t] = inflow[((long)b * 14 + 2 + tt) * LTOK + rest];
}

__global__ void embed_kernel(const float* __restrict__ x, const float* __restrict__ W,
                             const float* __restrict__ b, const float* __restrict__ pe,
                             float* __restrict__ out) {
    int t = blockIdx.x * blockDim.x + threadIdx.x;
    if (t >= MROWS * DMODEL) return;
    int n = t % DMODEL, row = t / DMODEL, l = row % LTOK;
    float s = b[n] + pe[l * DMODEL + n];
    const float* xr = x + (long)row * TLAG;
#pragma unroll
    for (int c = 0; c < TLAG; ++c) s += xr[c] * W[c * DMODEL + n];
    out[t] = s;
}

// ---------------- layernorm (row = 512) ----------------
__global__ __launch_bounds__(256)
void layernorm_k(const float* __restrict__ x, const float* __restrict__ res,
                 const float* __restrict__ g, const float* __restrict__ bb,
                 float* __restrict__ out) {
    int row = blockIdx.x;
    __shared__ float red[256];
    const float* xr = x + (long)row * DMODEL;
    const float* rr = res ? res + (long)row * DMODEL : nullptr;
    float v0 = xr[threadIdx.x]       + (rr ? rr[threadIdx.x]       : 0.0f);
    float v1 = xr[threadIdx.x + 256] + (rr ? rr[threadIdx.x + 256] : 0.0f);
    red[threadIdx.x] = v0 + v1; __syncthreads();
    for (int st = 128; st > 0; st >>= 1) { if (threadIdx.x < st) red[threadIdx.x] += red[threadIdx.x + st]; __syncthreads(); }
    float mu = red[0] / (float)DMODEL; __syncthreads();
    float d0 = v0 - mu, d1 = v1 - mu;
    red[threadIdx.x] = d0 * d0 + d1 * d1; __syncthreads();
    for (int st = 128; st > 0; st >>= 1) { if (threadIdx.x < st) red[threadIdx.x] += red[threadIdx.x + st]; __syncthreads(); }
    float inv = rsqrtf(red[0] / (float)DMODEL + 1e-5f);
    float* orow = out + (long)row * DMODEL;
    orow[threadIdx.x]       = d0 * inv * g[threadIdx.x]       + bb[threadIdx.x];
    orow[threadIdx.x + 256] = d1 * inv * g[threadIdx.x + 256] + bb[threadIdx.x + 256];
}

// ---------------- softmax over rows ----------------
__global__ __launch_bounds__(256)
void softmax_k(float* __restrict__ S, int ncols) {
    long row = blockIdx.x;
    float* r = S + row * (long)ncols;
    __shared__ float red[256];
    float mx = -1e30f;
    for (int i = threadIdx.x; i < ncols; i += 256) mx = fmaxf(mx, r[i]);
    red[threadIdx.x] = mx; __syncthreads();
    for (int st = 128; st > 0; st >>= 1) { if (threadIdx.x < st) red[threadIdx.x] = fmaxf(red[threadIdx.x], red[threadIdx.x + st]); __syncthreads(); }
    mx = red[0]; __syncthreads();
    float s = 0.0f;
    for (int i = threadIdx.x; i < ncols; i += 256) { float e = expf(r[i] - mx); r[i] = e; s += e; }
    red[threadIdx.x] = s; __syncthreads();
    for (int st = 128; st > 0; st >>= 1) { if (threadIdx.x < st) red[threadIdx.x] += red[threadIdx.x + st]; __syncthreads(); }
    float inv = 1.0f / red[0];
    for (int i = threadIdx.x; i < ncols; i += 256) r[i] *= inv;
}

// ---------------- ProbSparse attention pieces ----------------
__global__ void prob_measure(const float* __restrict__ qkv, const int* __restrict__ sidx,
                             float* __restrict__ Mbuf) {
    int t = blockIdx.x * blockDim.x + threadIdx.x;
    if (t >= NB * NHEAD * LTOK) return;
    int l = t % LTOK, h = (t / LTOK) % NHEAD, b = t / (LTOK * NHEAD);
    const float* q = qkv + ((long)(b * LTOK + l)) * QKVLD + h * DHEAD;
    float mx = -1e30f, sm = 0.0f;
    for (int s = 0; s < SK; ++s) {
        int ki = sidx[l * SK + s];
        const float* kr = qkv + ((long)(b * LTOK + ki)) * QKVLD + DMODEL + h * DHEAD;
        float d = 0.0f;
        for (int j = 0; j < DHEAD; ++j) d += q[j] * kr[j];
        mx = fmaxf(mx, d); sm += d;
    }
    Mbuf[t] = mx - sm / (float)LTOK;
}

__global__ __launch_bounds__(256)
void topk40(const float* __restrict__ Mbuf, int* __restrict__ top) {
    int bh = blockIdx.x;
    __shared__ float vals[LTOK];
    __shared__ float rv[256];
    __shared__ int   ri[256];
    const float* m = Mbuf + (long)bh * LTOK;
    for (int i = threadIdx.x; i < LTOK; i += 256) vals[i] = m[i];
    __syncthreads();
    for (int t = 0; t < SK; ++t) {
        float best = -1e30f; int bi = 0;
        for (int i = threadIdx.x; i < LTOK; i += 256)
            if (vals[i] > best) { best = vals[i]; bi = i; }
        rv[threadIdx.x] = best; ri[threadIdx.x] = bi; __syncthreads();
        for (int st = 128; st > 0; st >>= 1) {
            if (threadIdx.x < st && rv[threadIdx.x + st] > rv[threadIdx.x]) {
                rv[threadIdx.x] = rv[threadIdx.x + st]; ri[threadIdx.x] = ri[threadIdx.x + st];
            }
            __syncthreads();
        }
        if (threadIdx.x == 0) { top[bh * SK + t] = ri[0]; vals[ri[0]] = -1e30f; }
        __syncthreads();
    }
}

__global__ void vmean_k(const float* __restrict__ qkv, float* __restrict__ vm) {
    int t = blockIdx.x * blockDim.x + threadIdx.x;
    if (t >= NB * NHEAD * DHEAD) return;
    int d = t % DHEAD, h = (t / DHEAD) % NHEAD, b = t / (DHEAD * NHEAD);
    float s = 0.0f;
    for (int l = 0; l < LTOK; ++l)
        s += qkv[((long)(b * LTOK + l)) * QKVLD + 2 * DMODEL + h * DHEAD + d];
    vm[t] = s / (float)LTOK;
}

__global__ void fill_ctx(const float* __restrict__ vm, float* __restrict__ ctxh) {
    int t = blockIdx.x * blockDim.x + threadIdx.x;
    if (t >= NB * NHEAD * LTOK * DHEAD) return;
    int d = t % DHEAD;
    int bh = t / (DHEAD * LTOK);
    ctxh[t] = vm[bh * DHEAD + d];
}

__global__ void prob_scores(const float* __restrict__ qkv, const int* __restrict__ top,
                            float* __restrict__ sc) {
    int t = blockIdx.x * blockDim.x + threadIdx.x;
    if (t >= NB * NHEAD * SK * LTOK) return;
    int k = t % LTOK, u = (t / LTOK) % SK, bh = t / (LTOK * SK);
    int b = bh / NHEAD, h = bh % NHEAD;
    int qrow = top[bh * SK + u];
    const float* q  = qkv + ((long)(b * LTOK + qrow)) * QKVLD + h * DHEAD;
    const float* kr = qkv + ((long)(b * LTOK + k))    * QKVLD + DMODEL + h * DHEAD;
    float d = 0.0f;
    for (int j = 0; j < DHEAD; ++j) d += q[j] * kr[j];
    sc[t] = d * 0.125f;
}

__global__ void prob_ctx(const float* __restrict__ qkv, const int* __restrict__ top,
                         const float* __restrict__ sc, float* __restrict__ ctxh) {
    int t = blockIdx.x * blockDim.x + threadIdx.x;
    if (t >= NB * NHEAD * SK * DHEAD) return;
    int d = t % DHEAD, u = (t / DHEAD) % SK, bh = t / (DHEAD * SK);
    int b = bh / NHEAD, h = bh % NHEAD;
    int row = top[bh * SK + u];
    const float* p = sc + ((long)bh * SK + u) * LTOK;
    float s = 0.0f;
    for (int k = 0; k < LTOK; ++k)
        s += p[k] * qkv[((long)(b * LTOK + k)) * QKVLD + 2 * DMODEL + h * DHEAD + d];
    ctxh[((long)bh * LTOK + row) * DHEAD + d] = s;
}

__global__ void unheads_k(const float* __restrict__ ctxh, float* __restrict__ ctx2) {
    int t = blockIdx.x * blockDim.x + threadIdx.x;
    if (t >= MROWS * DMODEL) return;
    int c = t % DMODEL, l = (t / DMODEL) % LTOK, b = t / (DMODEL * LTOK);
    int h = c / DHEAD, d = c % DHEAD;
    ctx2[t] = ctxh[(((long)b * NHEAD + h) * LTOK + l) * DHEAD + d];
}

// ---------------- small scalar GEMMs at the tail ----------------
__global__ void proj_res(const float* __restrict__ dec, const float* __restrict__ W,
                         const float* __restrict__ b, const float* __restrict__ x,
                         float* __restrict__ po) {
    int t = blockIdx.x * blockDim.x + threadIdx.x;
    if (t >= MROWS * TLAG) return;
    int c = t % TLAG, row = t / TLAG;
    const float* dr = dec + (long)row * DMODEL;
    float s = b[c];
    for (int k = 0; k < DMODEL; ++k) s += dr[k] * W[k * TLAG + c];
    po[t] = s + x[t];
}

__global__ void fc_relu(const float* __restrict__ po, const float* __restrict__ W,
                        const float* __restrict__ b, float* __restrict__ h1) {
    int t = blockIdx.x * blockDim.x + threadIdx.x;
    if (t >= MROWS * 128) return;
    int n = t % 128, row = t / 128;
    const float* pr = po + (long)row * TLAG;
    float s = b[n];
#pragma unroll
    for (int c = 0; c < TLAG; ++c) s += pr[c] * W[c * 128 + n];
    h1[t] = fmaxf(s, 0.0f);
}

__global__ void fc_out(const float* __restrict__ h1, const float* __restrict__ W,
                       const float* __restrict__ b, float* __restrict__ o2) {
    int t = blockIdx.x * blockDim.x + threadIdx.x;
    if (t >= MROWS * 3) return;
    int n = t % 3, row = t / 3;
    const float* hr = h1 + (long)row * 128;
    float s = b[n];
    for (int c = 0; c < 128; ++c) s += hr[c] * W[c * 3 + n];
    o2[t] = s;
}

__global__ void final_reshape(const float* __restrict__ o2, float* __restrict__ out) {
    int t = blockIdx.x * blockDim.x + threadIdx.x;
    if (t >= NB * 3 * 40 * 38) return;
    int u = t % 38, s = (t / 38) % 40, p = (t / (38 * 40)) % 3, b = t / (38 * 40 * 3);
    int flat = p * LTOK + s * 38 + u;
    int l = flat / 3, c = flat % 3;
    out[t] = o2[((long)b * LTOK + l) * 3 + c];
}

// ---------------- host orchestration ----------------
static inline int cdiv(int a, int b) { return (a + b - 1) / b; }

extern "C" void kernel_launch(void* const* d_in, const int* in_sizes, int n_in,
                              void* d_out, int out_size, void* d_ws, size_t ws_size,
                              hipStream_t stream) {
    const float* inflow    = (const float*)d_in[4];
    const float* W_emb_e   = (const float*)d_in[5];
    const float* b_emb_e   = (const float*)d_in[6];
    const float* W_emb_d   = (const float*)d_in[7];
    const float* b_emb_d   = (const float*)d_in[8];
    const float* enc_Wqkv  = (const float*)d_in[9];
    const float* enc_bqkv  = (const float*)d_in[10];
    const float* enc_Wo    = (const float*)d_in[11];
    const float* enc_bo    = (const float*)d_in[12];
    const float* enc_ln1_g = (const float*)d_in[13];
    const float* enc_ln1_b = (const float*)d_in[14];
    const float* enc_Wf1   = (const float*)d_in[15];
    const float* enc_bf1   = (const float*)d_in[16];
    const float* enc_Wf2   = (const float*)d_in[17];
    const float* enc_bf2   = (const float*)d_in[18];
    const float* enc_ln2_g = (const float*)d_in[19];
    const float* enc_ln2_b = (const float*)d_in[20];
    const float* enc_norm_g= (const float*)d_in[21];
    const float* enc_norm_b= (const float*)d_in[22];
    const float* dec_Wqkv  = (const float*)d_in[23];
    const float* dec_bqkv  = (const float*)d_in[24];
    const float* dec_Wo_s  = (const float*)d_in[25];
    const float* dec_bo_s  = (const float*)d_in[26];
    const float* dec_ln1_g = (const float*)d_in[27];
    const float* dec_ln1_b = (const float*)d_in[28];
    const float* dec_Wq_c  = (const float*)d_in[29];
    const float* dec_bq_c  = (const float*)d_in[30];
    const float* dec_Wkv_c = (const float*)d_in[31];
    const float* dec_bkv_c = (const float*)d_in[32];
    const float* dec_Wo_c  = (const float*)d_in[33];
    const float* dec_bo_c  = (const float*)d_in[34];
    const float* dec_ln2_g = (const float*)d_in[35];
    const float* dec_ln2_b = (const float*)d_in[36];
    const float* dec_Wf1   = (const float*)d_in[37];
    const float* dec_bf1   = (const float*)d_in[38];
    const float* dec_Wf2   = (const float*)d_in[39];
    const float* dec_bf2   = (const float*)d_in[40];
    const float* dec_ln3_g = (const float*)d_in[41];
    const float* dec_ln3_b = (const float*)d_in[42];
    const float* W_proj    = (const float*)d_in[43];
    const float* b_proj    = (const float*)d_in[44];
    const float* W1        = (const float*)d_in[45];
    const float* b1        = (const float*)d_in[46];
    const float* W3        = (const float*)d_in[47];
    const float* b3        = (const float*)d_in[48];
    const int*   samp      = (const int*)d_in[49];
    float* OUT = (float*)d_out;

    // ---- workspace layout (floats) ----
    float* W = (float*)d_ws;
    size_t off = 0;
    auto alloc = [&](size_t n) { float* p = W + off; off += n; return p; };
    float* pe     = alloc((size_t)LTOK * DMODEL);
    float* xbuf   = alloc((size_t)MROWS * TLAG);
    float* enc    = alloc((size_t)MROWS * DMODEL);
    float* dec    = alloc((size_t)MROWS * DMODEL);
    float* big    = alloc((size_t)MROWS * DFFN);     // union: qkv / ffn hidden
    float* ctxh   = alloc((size_t)NB * NHEAD * LTOK * DHEAD);
    float* ctx2   = alloc((size_t)MROWS * DMODEL);
    float* tmp    = alloc((size_t)MROWS * DMODEL);
    float* qc     = alloc((size_t)MROWS * DMODEL);
    float* kvc    = alloc((size_t)MROWS * 2 * DMODEL);
    float* Mbuf   = alloc((size_t)NB * NHEAD * LTOK);
    float* vmeanb = alloc((size_t)NB * NHEAD * DHEAD);
    float* scP    = alloc((size_t)NB * NHEAD * SK * LTOK);
    float* scX    = alloc((size_t)LTOK * LTOK);
    float* po     = alloc((size_t)MROWS * TLAG);
    float* h1     = alloc((size_t)MROWS * 128);
    float* o2     = alloc((size_t)MROWS * 3);
    int*   topb   = (int*)alloc((size_t)NB * NHEAD * SK);
    (void)ws_size; (void)in_sizes; (void)n_in; (void)out_size;

    const int T = 256;
    auto g1 = [&](int n) { return dim3(cdiv(n, T)); };

    pe_kernel<<<g1(LTOK * DMODEL), T, 0, stream>>>(pe);
    build_x<<<g1(MROWS * TLAG), T, 0, stream>>>(inflow, xbuf);

    auto gemm = [&](const float* A, int lda, const float* Wt, int ldw, int transB,
                    const float* bias, float* C, int ldc, int M, int N, int K,
                    int act, float scale) {
        dim3 grid(cdiv(N, BN), cdiv(M, BM));
        bool exact = (M % BM == 0) && (N % BN == 0) && (K % BK == 0);
        if (exact)
            gemm_wmma<1><<<grid, 256, 0, stream>>>(A, lda, Wt, ldw, transB, bias, C, ldc, M, N, K, act, scale);
        else
            gemm_wmma<0><<<grid, 256, 0, stream>>>(A, lda, Wt, ldw, transB, bias, C, ldc, M, N, K, act, scale);
    };

    auto prob_attn_block = [&](float* io, const float* Wqkv, const float* bqkv,
                               const float* Wo, const float* bo,
                               const float* lng, const float* lnb, const int* sidx) {
        gemm(io, DMODEL, Wqkv, QKVLD, 0, bqkv, big, QKVLD, MROWS, QKVLD, DMODEL, 0, 1.0f);
        prob_measure<<<g1(NB * NHEAD * LTOK), T, 0, stream>>>(big, sidx, Mbuf);
        topk40<<<NB * NHEAD, 256, 0, stream>>>(Mbuf, topb);
        vmean_k<<<g1(NB * NHEAD * DHEAD), T, 0, stream>>>(big, vmeanb);
        fill_ctx<<<g1(NB * NHEAD * LTOK * DHEAD), T, 0, stream>>>(vmeanb, ctxh);
        prob_scores<<<g1(NB * NHEAD * SK * LTOK), T, 0, stream>>>(big, topb, scP);
        softmax_k<<<NB * NHEAD * SK, 256, 0, stream>>>(scP, LTOK);
        prob_ctx<<<g1(NB * NHEAD * SK * DHEAD), T, 0, stream>>>(big, topb, scP, ctxh);
        unheads_k<<<g1(MROWS * DMODEL), T, 0, stream>>>(ctxh, ctx2);
        gemm(ctx2, DMODEL, Wo, DMODEL, 0, bo, tmp, DMODEL, MROWS, DMODEL, DMODEL, 0, 1.0f);
        layernorm_k<<<MROWS, 256, 0, stream>>>(io, tmp, lng, lnb, io);
    };

    auto ffn_block = [&](float* io, const float* Wf1, const float* bf1,
                         const float* Wf2, const float* bf2,
                         const float* lng, const float* lnb) {
        gemm(io, DMODEL, Wf1, DFFN, 0, bf1, big, DFFN, MROWS, DFFN, DMODEL, 1, 1.0f);
        gemm(big, DFFN, Wf2, DMODEL, 0, bf2, tmp, DMODEL, MROWS, DMODEL, DFFN, 0, 1.0f);
        layernorm_k<<<MROWS, 256, 0, stream>>>(io, tmp, lng, lnb, io);
    };

    // ---- encoder ----
    embed_kernel<<<g1(MROWS * DMODEL), T, 0, stream>>>(xbuf, W_emb_e, b_emb_e, pe, enc);
    for (int i = 0; i < 2; ++i) {
        prob_attn_block(enc,
                        enc_Wqkv + (size_t)i * DMODEL * QKVLD, enc_bqkv + (size_t)i * QKVLD,
                        enc_Wo   + (size_t)i * DMODEL * DMODEL, enc_bo + (size_t)i * DMODEL,
                        enc_ln1_g + (size_t)i * DMODEL, enc_ln1_b + (size_t)i * DMODEL,
                        samp + (size_t)i * LTOK * SK);
        ffn_block(enc,
                  enc_Wf1 + (size_t)i * DMODEL * DFFN, enc_bf1 + (size_t)i * DFFN,
                  enc_Wf2 + (size_t)i * DFFN * DMODEL, enc_bf2 + (size_t)i * DMODEL,
                  enc_ln2_g + (size_t)i * DMODEL, enc_ln2_b + (size_t)i * DMODEL);
    }
    layernorm_k<<<MROWS, 256, 0, stream>>>(enc, nullptr, enc_norm_g, enc_norm_b, enc);

    // ---- decoder ----
    embed_kernel<<<g1(MROWS * DMODEL), T, 0, stream>>>(xbuf, W_emb_d, b_emb_d, pe, dec);
    prob_attn_block(dec, dec_Wqkv, dec_bqkv, dec_Wo_s, dec_bo_s,
                    dec_ln1_g, dec_ln1_b, samp + (size_t)2 * LTOK * SK);

    gemm(dec, DMODEL, dec_Wq_c, DMODEL, 0, dec_bq_c, qc, DMODEL, MROWS, DMODEL, DMODEL, 0, 1.0f);
    gemm(enc, DMODEL, dec_Wkv_c, 2 * DMODEL, 0, dec_bkv_c, kvc, 2 * DMODEL, MROWS, 2 * DMODEL, DMODEL, 0, 1.0f);
    for (int b = 0; b < NB; ++b) {
        for (int h = 0; h < NHEAD; ++h) {
            const float* Qb = qc  + ((size_t)b * LTOK) * DMODEL + h * DHEAD;
            const float* Kb = kvc + ((size_t)b * LTOK) * 2 * DMODEL + h * DHEAD;
            const float* Vb = kvc + ((size_t)b * LTOK) * 2 * DMODEL + DMODEL + h * DHEAD;
            gemm(Qb, DMODEL, Kb, 2 * DMODEL, 1, nullptr, scX, LTOK, LTOK, LTOK, DHEAD, 0, 0.125f);
            softmax_k<<<LTOK, 256, 0, stream>>>(scX, LTOK);
            gemm(scX, LTOK, Vb, 2 * DMODEL, 0, nullptr,
                 ctxh + (((size_t)b * NHEAD + h) * LTOK) * DHEAD, DHEAD,
                 LTOK, DHEAD, LTOK, 0, 1.0f);
        }
    }
    unheads_k<<<g1(MROWS * DMODEL), T, 0, stream>>>(ctxh, ctx2);
    gemm(ctx2, DMODEL, dec_Wo_c, DMODEL, 0, dec_bo_c, tmp, DMODEL, MROWS, DMODEL, DMODEL, 0, 1.0f);
    layernorm_k<<<MROWS, 256, 0, stream>>>(dec, tmp, dec_ln2_g, dec_ln2_b, dec);
    ffn_block(dec, dec_Wf1, dec_bf1, dec_Wf2, dec_bf2, dec_ln3_g, dec_ln3_b);

    proj_res<<<g1(MROWS * TLAG), T, 0, stream>>>(dec, W_proj, b_proj, xbuf, po);
    fc_relu<<<g1(MROWS * 128), T, 0, stream>>>(po, W1, b1, h1);
    fc_out<<<g1(MROWS * 3), T, 0, stream>>>(h1, W3, b3, o2);
    final_reshape<<<g1(NB * 3 * 40 * 38), T, 0, stream>>>(o2, OUT);
}
